// MultiHeadAttention_47553877901548
// MI455X (gfx1250) — matrix operand order repaired and checked by reading, not compiled, not run on hardware
//
#include <hip/hip_runtime.h>
#include <hip/hip_bf16.h>

// ---------------------------------------------------------------------------
// MHA for MI455X (gfx1250, wave32). Compute-bound -> everything on the WMMA
// pipe (V_WMMA_F32_16X16X32_F16, f16 in / f32 accum). Intermediates are f16
// in workspace (L2-resident: 192MB covers all ~80MB of them).
// GEMM B-tiles staged to LDS with GLOBAL_LOAD_ASYNC_TO_LDS_B128 (ASYNCcnt,
// double buffered). All fragment loads are contiguous-per-lane b128s.
// ---------------------------------------------------------------------------

typedef _Float16 f16;
typedef __attribute__((ext_vector_type(16))) _Float16 v16h;
typedef __attribute__((ext_vector_type(8)))  _Float16 v8h;
typedef __attribute__((ext_vector_type(4)))  _Float16 v4h;
typedef __attribute__((ext_vector_type(8)))  float    v8f;

#define DIMC   1024
#define NHEAD  16
#define HDIM   64
#define BATCH  4
#define SEQ    2048
#define NTOK   (BATCH * SEQ)          // 8192
#define SCALE_F 0.125f                // 64^-0.5

// ---- WMMA wrapper ---------------------------------------------------------
__device__ __forceinline__ v8f wmma_f16(v16h a, v16h b, v8f c) {
  return __builtin_amdgcn_wmma_f32_16x16x32_f16(
      false, a, false, b, (short)0, c, false, false);
}

// ---- Fragment loaders (CDNA5 ISA 7.12.2 layouts, wave32) ------------------
// A 16x32 (MxK): lane half g holds K = 8g+{0..7} and 16+8g+{0..7}; row = lane&15.
// Per lane: two contiguous 16B runs -> two b128 loads.
__device__ __forceinline__ v16h load_a_frag(const f16* __restrict__ base, int ld) {
  const int lane = threadIdx.x & 31;
  const int r = lane & 15, g = lane >> 4;
  const f16* p = base + (size_t)r * ld + g * 8;
  v8h lo = *(const v8h*)(p);
  v8h hi = *(const v8h*)(p + 16);
  return __builtin_shufflevector(lo, hi, 0,1,2,3,4,5,6,7,8,9,10,11,12,13,14,15);
}

// B 32x16 (KxN) where memory holds B^T row-major: B[k][n] = base[n*ld + k].
// Per lane: K = 8g+{0..7} and 16+8g+{0..7} -> two b128 loads.
__device__ __forceinline__ v16h load_b_frag_trans(const f16* __restrict__ base, int ld) {
  const int lane = threadIdx.x & 31;
  const int n = lane & 15, g = lane >> 4;
  const f16* p = base + (size_t)n * ld + g * 8;
  v8h lo = *(const v8h*)(p);
  v8h hi = *(const v8h*)(p + 16);
  return __builtin_shufflevector(lo, hi, 0,1,2,3,4,5,6,7,8,9,10,11,12,13,14,15);
}

// ---- Async copy: one wave-wide instruction moves 32 lanes x 16B -----------
__device__ __forceinline__ void async_load_b128(unsigned lds_off, const f16* gaddr) {
  asm volatile("global_load_async_to_lds_b128 %0, %1, off"
               :: "v"(lds_off), "v"(gaddr) : "memory");
}

// ---- f32 -> f16 conversion (vectorized: float4 in, 4xf16 out) -------------
__global__ void cvt_f16_kernel(const float* __restrict__ src, f16* __restrict__ dst, int n4) {
  int i = blockIdx.x * blockDim.x + threadIdx.x;
  if (i < n4) {
    const float4 v = ((const float4*)src)[i];
    v4h o;
    o[0] = (f16)v.x; o[1] = (f16)v.y; o[2] = (f16)v.z; o[3] = (f16)v.w;
    ((v4h*)dst)[i] = o;
  }
}

// ---- GEMM: out[M=8192, N=1024] = X(f16) @ W^T, W row-major [n][k] f16 -----
// 8 waves; wave owns a 32x64 tile (2 A-frags x 4 B-frags = 8 WMMA / K-step).
// Shared B tile (64 n-rows x 32 k) staged in LDS via async loads, double buf.
// MODE: 0 = f16 [tok][dim], 1 = f16 V-transposed per head [b,h][d][tok], 2 = f32.
template <int MODE>
__global__ __launch_bounds__(256) void wmma_gemm_xwt(
    const f16* __restrict__ X, const f16* __restrict__ W,
    f16* __restrict__ outH, float* __restrict__ outF) {
  __shared__ f16 bstage[2][64 * 32];            // 2 x 4KB

  const int tid  = threadIdx.x;
  const int lane = tid & 31;
  const int wid  = tid >> 5;
  const int r = lane & 15, g = lane >> 4;
  const int m0 = (blockIdx.x * 8 + wid) * 32;   // 32 rows per wave
  const int n0 = blockIdx.y * 64;

  // async stage helper indices: 256 threads x 16B = 4KB tile
  const int srow = tid >> 2;                    // 0..63  (n-local)
  const int sch  = (tid & 3) * 8;               // half-offset within k-row

  // prologue: stage k0 = 0 into buffer 0
  async_load_b128((unsigned)(uintptr_t)&bstage[0][srow * 32 + sch],
                  W + (size_t)(n0 + srow) * DIMC + sch);

  v8f acc[2][4] = {};
  int buf = 0;
  for (int k0 = 0; k0 < DIMC; k0 += 32, buf ^= 1) {
    const bool more = (k0 + 32) < DIMC;
    if (more)
      async_load_b128((unsigned)(uintptr_t)&bstage[buf ^ 1][srow * 32 + sch],
                      W + (size_t)(n0 + srow) * DIMC + (k0 + 32) + sch);
    if (more) asm volatile("s_wait_asynccnt 0x1" ::: "memory");
    else      asm volatile("s_wait_asynccnt 0x0" ::: "memory");
    __syncthreads();

    const v16h a0 = load_a_frag(X + (size_t)m0 * DIMC + k0, DIMC);
    const v16h a1 = load_a_frag(X + (size_t)(m0 + 16) * DIMC + k0, DIMC);
#pragma unroll
    for (int t = 0; t < 4; ++t) {
      const v16h b = load_b_frag_trans(&bstage[buf][16 * t * 32], 32);
      acc[0][t] = wmma_f16(a0, b, acc[0][t]);
      acc[1][t] = wmma_f16(a1, b, acc[1][t]);
    }
    __syncthreads();   // protect bstage[buf] before it is re-staged
  }

#pragma unroll
  for (int mi = 0; mi < 2; ++mi) {
#pragma unroll
    for (int t = 0; t < 4; ++t) {
#pragma unroll
      for (int i = 0; i < 8; ++i) {
        const int m = m0 + mi * 16 + i + 8 * g;   // token index
        const int n = n0 + 16 * t + r;            // dim index
        if constexpr (MODE == 2) {
          outF[(size_t)m * DIMC + n] = acc[mi][t][i];
        } else if constexpr (MODE == 1) {
          const int bq = m >> 11, q = m & (SEQ - 1);
          const int h = n >> 6, dh = n & (HDIM - 1);
          outH[(((size_t)bq * NHEAD + h) * HDIM + dh) * SEQ + q] = (f16)acc[mi][t][i];
        } else {
          outH[(size_t)m * DIMC + n] = (f16)acc[mi][t][i];
        }
      }
    }
  }
}

// ---- Flash attention (causal). Block = one (b,h) x 64 queries, 4 waves. ---
// Q,K: [B,N,H*d] (row stride DIM).  Vt: per-head transposed [b,h][d][n].
__global__ __launch_bounds__(128) void flash_attn_kernel(
    const f16* __restrict__ Q, const f16* __restrict__ K,
    const f16* __restrict__ Vt, f16* __restrict__ O) {
  __shared__ f16 ldsP[4][16 * 64];   // per-wave P staging (C-frag -> A-frag)

  const int lane = threadIdx.x & 31;
  const int wid  = threadIdx.x >> 5;
  const int r = lane & 15, g = lane >> 4;
  const int b = blockIdx.x >> 4;
  const int h = blockIdx.x & 15;
  const int qbase = blockIdx.y * 64 + wid * 16;

  const size_t headOff = ((size_t)b * SEQ) * DIMC + (size_t)h * HDIM;
  const f16* Qp = Q + headOff;
  const f16* Kp = K + headOff;
  const f16* Vp = Vt + ((size_t)(b * NHEAD + h) * HDIM) * SEQ;  // [d][n]
  f16* pbuf = &ldsP[wid][0];

  const v16h qa0 = load_a_frag(Qp + (size_t)qbase * DIMC, DIMC);
  const v16h qa1 = load_a_frag(Qp + (size_t)qbase * DIMC + 32, DIMC);

  float mrow[8], lrow[8];
#pragma unroll
  for (int i = 0; i < 8; ++i) { mrow[i] = -1e30f; lrow[i] = 0.f; }
  v8f o[4] = {};

  for (int kb = 0; kb <= qbase + 15; kb += 64) {
    // S = (Q K^T) * scale
    v8f s[4];
#pragma unroll
    for (int t = 0; t < 4; ++t) {
      v8f sc = {};
      sc = wmma_f16(qa0, load_b_frag_trans(Kp + (size_t)(kb + 16 * t) * DIMC, DIMC), sc);
      sc = wmma_f16(qa1, load_b_frag_trans(Kp + (size_t)(kb + 16 * t) * DIMC + 32, DIMC), sc);
#pragma unroll
      for (int i = 0; i < 8; ++i) s[t][i] = sc[i] * SCALE_F;
    }
    // causal mask (only tiles overlapping the diagonal)
    if (kb + 63 > qbase) {
#pragma unroll
      for (int t = 0; t < 4; ++t)
#pragma unroll
        for (int i = 0; i < 8; ++i) {
          if (kb + 16 * t + r > qbase + i + 8 * g) s[t][i] = -1e30f;
        }
    }
    // row max (row lives in one 16-lane half)
    float tmax[8];
#pragma unroll
    for (int i = 0; i < 8; ++i)
      tmax[i] = fmaxf(fmaxf(s[0][i], s[1][i]), fmaxf(s[2][i], s[3][i]));
#pragma unroll
    for (int off = 1; off < 16; off <<= 1)
#pragma unroll
      for (int i = 0; i < 8; ++i)
        tmax[i] = fmaxf(tmax[i], __shfl_xor(tmax[i], off, 32));

    float corr[8], rsum[8];
#pragma unroll
    for (int i = 0; i < 8; ++i) {
      const float nm = fmaxf(mrow[i], tmax[i]);
      corr[i] = __expf(mrow[i] - nm);
      mrow[i] = nm;
      rsum[i] = 0.f;
    }
#pragma unroll
    for (int t = 0; t < 4; ++t)
#pragma unroll
      for (int i = 0; i < 8; ++i) {
        const float p = __expf(s[t][i] - mrow[i]);
        rsum[i] += p;
        pbuf[(i + 8 * g) * 64 + 16 * t + r] = (f16)p;
      }
#pragma unroll
    for (int off = 1; off < 16; off <<= 1)
#pragma unroll
      for (int i = 0; i < 8; ++i)
        rsum[i] += __shfl_xor(rsum[i], off, 32);
#pragma unroll
    for (int i = 0; i < 8; ++i) lrow[i] = lrow[i] * corr[i] + rsum[i];
#pragma unroll
    for (int t = 0; t < 4; ++t)
#pragma unroll
      for (int i = 0; i < 8; ++i) o[t][i] *= corr[i];

    // LDS write -> read hazard (wave-private buffer, DS is wave-in-order)
    asm volatile("s_wait_dscnt 0" ::: "memory");

    // O += P @ V  (keys split 32+32; V pre-transposed -> contiguous B loads)
    const v16h pa0 = load_a_frag(pbuf, 64);
    const v16h pa1 = load_a_frag(pbuf + 32, 64);
#pragma unroll
    for (int t = 0; t < 4; ++t) {
      o[t] = wmma_f16(pa0, load_b_frag_trans(Vp + (size_t)(16 * t) * SEQ + kb, SEQ), o[t]);
      o[t] = wmma_f16(pa1, load_b_frag_trans(Vp + (size_t)(16 * t) * SEQ + kb + 32, SEQ), o[t]);
    }
  }

  float inv[8];
#pragma unroll
  for (int i = 0; i < 8; ++i) inv[i] = 1.0f / lrow[i];
  f16* Op = O + headOff;
#pragma unroll
  for (int t = 0; t < 4; ++t)
#pragma unroll
    for (int i = 0; i < 8; ++i)
      Op[(size_t)(qbase + i + 8 * g) * DIMC + 16 * t + r] = (f16)(o[t][i] * inv[i]);
}

// ---------------------------------------------------------------------------
extern "C" void kernel_launch(void* const* d_in, const int* in_sizes, int n_in,
                              void* d_out, int out_size, void* d_ws, size_t ws_size,
                              hipStream_t stream) {
  (void)in_sizes; (void)n_in; (void)out_size; (void)ws_size;
  const float* x  = (const float*)d_in[0];
  // d_in[1]: causal bool mask -> implied by loop bounds, unused.
  const float* Wq = (const float*)d_in[2];
  const float* Wk = (const float*)d_in[3];
  const float* Wv = (const float*)d_in[4];
  const float* Wo = (const float*)d_in[5];
  float* out = (float*)d_out;

  // Workspace (f16): Xh 16MB | Wq/Wk/Wv/Wo 4x2MB | Q 16 | K 16 | Vt 16 | A 16
  char* ws = (char*)d_ws;
  f16* Xh  = (f16*)ws;
  f16* Wqh = (f16*)(ws + (size_t)16 * 1024 * 1024);
  f16* Wkh = Wqh + (size_t)DIMC * DIMC;
  f16* Wvh = Wkh + (size_t)DIMC * DIMC;
  f16* Woh = Wvh + (size_t)DIMC * DIMC;
  f16* Qh  = (f16*)(ws + (size_t)24 * 1024 * 1024);
  f16* Kh  = Qh + (size_t)NTOK * DIMC;
  f16* Vth = Kh + (size_t)NTOK * DIMC;
  f16* Ah  = Vth + (size_t)NTOK * DIMC;

  const int nx4 = (NTOK * DIMC) / 4;
  cvt_f16_kernel<<<(nx4 + 255) / 256, 256, 0, stream>>>(x, Xh, nx4);
  const int nw4 = (DIMC * DIMC) / 4;
  cvt_f16_kernel<<<(nw4 + 255) / 256, 256, 0, stream>>>(Wq, Wqh, nw4);
  cvt_f16_kernel<<<(nw4 + 255) / 256, 256, 0, stream>>>(Wk, Wkh, nw4);
  cvt_f16_kernel<<<(nw4 + 255) / 256, 256, 0, stream>>>(Wv, Wvh, nw4);
  cvt_f16_kernel<<<(nw4 + 255) / 256, 256, 0, stream>>>(Wo, Woh, nw4);

  dim3 gemm_grid(NTOK / 256, DIMC / 64);      // 32 x 16
  wmma_gemm_xwt<0><<<gemm_grid, 256, 0, stream>>>(Xh, Wqh, Qh, nullptr);
  wmma_gemm_xwt<0><<<gemm_grid, 256, 0, stream>>>(Xh, Wkh, Kh, nullptr);
  wmma_gemm_xwt<1><<<gemm_grid, 256, 0, stream>>>(Xh, Wvh, Vth, nullptr);

  flash_attn_kernel<<<dim3(BATCH * NHEAD, SEQ / 64), 128, 0, stream>>>(Qh, Kh, Vth, Ah);

  wmma_gemm_xwt<2><<<gemm_grid, 256, 0, stream>>>(Ah, Woh, nullptr, out);
}
